// OurLoRAT5DenseActDense_31267361915293
// MI455X (gfx1250) — compile-verified, matrix operand order checked by reading, not run on hardware
//
#include <hip/hip_runtime.h>
#include <hip/hip_bf16.h>

// ---------------------------------------------------------------------------
// LoRA T5 dense-act-dense for MI455X (gfx1250), bf16 WMMA path.
// All matmuls (main GEMMs *and* rank-32 LoRA projections) run on
// v_wmma_f32_16x16x32_bf16. Main GEMM uses double-buffered LDS with an
// async-global->LDS fast path (ASYNCcnt) when the toolchain exposes it.
// ---------------------------------------------------------------------------

typedef __attribute__((ext_vector_type(16))) __bf16 v16bf;
typedef __attribute__((ext_vector_type(8)))  float  v8f;
typedef int v4i_gcc __attribute__((vector_size(16)));   // matches builtin param

#define ALPHA_ 2.0f
#define BATCH_ 32
#define SEQ_   512
#define DIM_   1024
#define FF_    4096
#define MROWS_ (BATCH_ * SEQ_)   // 16384
#define NEXP_  8

#define BM 128
#define BN 128
#define BK 64
#define LDSS 72   // LDS row stride in bf16 elems (144 B = 9*16 B, keeps b128 aligned)

#if defined(__gfx1250__) && __has_builtin(__builtin_amdgcn_global_load_async_to_lds_b128)
#define HAVE_ASYNC_LDS 1
#endif

__device__ __forceinline__ unsigned short f2bf(float f) {
  unsigned u = __builtin_bit_cast(unsigned, f);
  unsigned r = (u + 0x7FFFu + ((u >> 16) & 1u)) >> 16;   // round-nearest-even
  return (unsigned short)r;
}

union Frag { v16bf v; uint4 q[2]; };

#ifdef HAVE_ASYNC_LDS
__device__ __forceinline__ void async_copy16(const unsigned short* g, unsigned short* l) {
  // Signature (from hipcc diagnostic): param0 = AS1 int4*, param1 = AS3 int4*.
  __builtin_amdgcn_global_load_async_to_lds_b128(
      (__attribute__((address_space(1))) v4i_gcc*)g,
      (__attribute__((address_space(3))) v4i_gcc*)l, 0, 0);
}
__device__ __forceinline__ void async_wait0() {
#if __has_builtin(__builtin_amdgcn_s_wait_asynccnt)
  __builtin_amdgcn_s_wait_asynccnt(0);
#else
  asm volatile("s_wait_asynccnt 0x0" ::: "memory");
#endif
}
#endif

// ---------------------------------------------------------------------------
// Elementwise f32 -> bf16 conversion.
// ---------------------------------------------------------------------------
__global__ void cvt_bf16_kernel(const float* __restrict__ in,
                                unsigned short* __restrict__ out, long n) {
  long i = (long)blockIdx.x * blockDim.x + threadIdx.x;
  long stride = (long)gridDim.x * blockDim.x;
  for (; i < n; i += stride) out[i] = f2bf(in[i]);
}

// ---------------------------------------------------------------------------
// WMMA low-rank projection:  low[M,32] = ALPHA * X[M,K] @ [ca ; ea[e]]^T
// (skinny GEMM, N=32). A-fragments come straight from global: each X row is
// consumed by exactly one wave, so LDS staging has no reuse to exploit; the
// 32xK B operand is tiny and L2/L0 resident. 8 waves/block, 16 rows/wave.
// ---------------------------------------------------------------------------
__global__ __launch_bounds__(256)
void lowrank_wmma_kernel(const unsigned short* __restrict__ X,  // [M,K] bf16
                         const unsigned short* __restrict__ ca, // [16,K] bf16
                         const unsigned short* __restrict__ ea, // [E,16,K] bf16
                         const int* __restrict__ eids,
                         unsigned short* __restrict__ low,      // [M,32] bf16
                         int K) {
  const int tid  = threadIdx.x;
  const int lane = tid & 31;
  const int wave = tid >> 5;
  const int row0 = blockIdx.x * 128 + wave * 16;   // block spans one batch row
  const int e    = eids[(blockIdx.x * 128) / SEQ_];

  const int lrow = lane & 15;
  const int lk   = (lane >> 4) * 8;

  v8f acc0 = {}, acc1 = {};
  const unsigned short* xr  = X  + (long)(row0 + lrow) * K;
  const unsigned short* car = ca + (long)lrow * K;
  const unsigned short* ear = ea + ((long)e * 16 + lrow) * K;

  for (int k0 = 0; k0 < K; k0 += 32) {
    Frag af, b0, b1;
    af.q[0] = *reinterpret_cast<const uint4*>(xr + k0 + lk);
    af.q[1] = *reinterpret_cast<const uint4*>(xr + k0 + lk + 16);
    b0.q[0] = *reinterpret_cast<const uint4*>(car + k0 + lk);
    b0.q[1] = *reinterpret_cast<const uint4*>(car + k0 + lk + 16);
    b1.q[0] = *reinterpret_cast<const uint4*>(ear + k0 + lk);
    b1.q[1] = *reinterpret_cast<const uint4*>(ear + k0 + lk + 16);
    acc0 = __builtin_amdgcn_wmma_f32_16x16x32_bf16(false, af.v, false, b0.v,
                                                   (short)0, acc0, false, false);
    acc1 = __builtin_amdgcn_wmma_f32_16x16x32_bf16(false, af.v, false, b1.v,
                                                   (short)0, acc1, false, false);
  }

  const int crow = (lane >> 4) * 8;
  const int ccol = lane & 15;
  #pragma unroll
  for (int v = 0; v < 8; ++v) {
    long r = row0 + crow + v;
    low[r * 32 + ccol]      = f2bf(ALPHA_ * acc0[v]);
    low[r * 32 + 16 + ccol] = f2bf(ALPHA_ * acc1[v]);
  }
}

// ---------------------------------------------------------------------------
// Tiled WMMA GEMM:  C[M,N] = A[M,K] @ Bt[N,K]^T  (+ rank-32 LoRA step)
// 256 threads = 8 waves; block tile 128x128; wave tile 32x64 (2x4 WMMA tiles);
// BK=64 (two k-steps, 16 WMMA per wave per barrier); double-buffered LDS.
// ---------------------------------------------------------------------------
template <bool RELU, typename OutT>
__global__ __launch_bounds__(256)
void gemm_lora_kernel(const unsigned short* __restrict__ A,     // [M,K] bf16
                      const unsigned short* __restrict__ Bt,    // [N,K] bf16
                      const unsigned short* __restrict__ lowA,  // [M,32] bf16 (ALPHA folded)
                      const unsigned short* __restrict__ lowBc, // [N,16] bf16
                      const unsigned short* __restrict__ lowBe, // [E,N,16] bf16
                      const int* __restrict__ eids,
                      OutT* __restrict__ C, int M, int N, int K) {
  __shared__ unsigned short sA[2][BM * LDSS];
  __shared__ unsigned short sB[2][BN * LDSS];

  const int tid  = threadIdx.x;
  const int lane = tid & 31;
  const int wave = tid >> 5;
  const int wm   = wave & 3;                 // 4 waves along M (32 rows each)
  const int wn   = wave >> 2;                // 2 waves along N (64 cols each)
  const int bm   = blockIdx.y * BM;
  const int bn   = blockIdx.x * BN;
  const int e    = eids[bm / SEQ_];          // block tile lies in one batch row

  v8f acc[2][4];
  #pragma unroll
  for (int i = 0; i < 2; ++i)
    #pragma unroll
    for (int j = 0; j < 4; ++j)
      acc[i][j] = {};

  const int lrow = lane & 15;                // fragment row within 16
  const int lk   = (lane >> 4) * 8;          // K sub-offset: 0 or 8 (ISA layout)
  const int nk   = K / BK;

  // Per-thread staging coordinates: 4 chunks of 16 B per tile (A and B each).
  // chunk = tid*4+i : row = chunk>>3 (0..127), co = (chunk&7)*8 (0..56).
#ifdef HAVE_ASYNC_LDS
  auto issue_tile = [&](int k0, int buf) {
    #pragma unroll
    for (int i = 0; i < 4; ++i) {
      int chunk = tid * 4 + i;
      int row   = chunk >> 3;
      int co    = (chunk & 7) * 8;
      async_copy16(A  + (long)(bm + row) * K + k0 + co, &sA[buf][row * LDSS + co]);
      async_copy16(Bt + (long)(bn + row) * K + k0 + co, &sB[buf][row * LDSS + co]);
    }
  };
  issue_tile(0, 0);
  async_wait0();
  __syncthreads();
#else
  uint4 ra[4], rb[4];
  auto g2r = [&](int k0) {
    #pragma unroll
    for (int i = 0; i < 4; ++i) {
      int chunk = tid * 4 + i;
      int row   = chunk >> 3;
      int co    = (chunk & 7) * 8;
      ra[i] = *reinterpret_cast<const uint4*>(A  + (long)(bm + row) * K + k0 + co);
      rb[i] = *reinterpret_cast<const uint4*>(Bt + (long)(bn + row) * K + k0 + co);
    }
  };
  auto r2s = [&](int buf) {
    #pragma unroll
    for (int i = 0; i < 4; ++i) {
      int chunk = tid * 4 + i;
      int row   = chunk >> 3;
      int co    = (chunk & 7) * 8;
      *reinterpret_cast<uint4*>(&sA[buf][row * LDSS + co]) = ra[i];
      *reinterpret_cast<uint4*>(&sB[buf][row * LDSS + co]) = rb[i];
    }
  };
  g2r(0);
  r2s(0);
  __syncthreads();
#endif

  for (int k = 0; k < nk; ++k) {
    const int cur = k & 1;
    const int nxt = cur ^ 1;

#ifdef HAVE_ASYNC_LDS
    if (k + 1 < nk) issue_tile((k + 1) * BK, nxt);   // overlap with compute
#else
    if (k + 1 < nk) g2r((k + 1) * BK);               // global loads in flight
#endif

    // Compute the two 32-wide k-steps of this BK=64 block.
    #pragma unroll
    for (int kk = 0; kk < 2; ++kk) {
      const int ko = kk * 32;
      Frag af[2], bf4[4];
      #pragma unroll
      for (int tm = 0; tm < 2; ++tm) {
        int r = wm * 32 + tm * 16 + lrow;
        af[tm].q[0] = *reinterpret_cast<const uint4*>(&sA[cur][r * LDSS + ko + lk]);
        af[tm].q[1] = *reinterpret_cast<const uint4*>(&sA[cur][r * LDSS + ko + lk + 16]);
      }
      #pragma unroll
      for (int tn = 0; tn < 4; ++tn) {
        int r = wn * 64 + tn * 16 + lrow;
        bf4[tn].q[0] = *reinterpret_cast<const uint4*>(&sB[cur][r * LDSS + ko + lk]);
        bf4[tn].q[1] = *reinterpret_cast<const uint4*>(&sB[cur][r * LDSS + ko + lk + 16]);
      }
      #pragma unroll
      for (int tm = 0; tm < 2; ++tm)
        #pragma unroll
        for (int tn = 0; tn < 4; ++tn)
          acc[tm][tn] = __builtin_amdgcn_wmma_f32_16x16x32_bf16(
              false, af[tm].v, false, bf4[tn].v, (short)0, acc[tm][tn],
              false, false);
    }

#ifdef HAVE_ASYNC_LDS
    if (k + 1 < nk) async_wait0();                   // next buffer resident
#else
    if (k + 1 < nk) r2s(nxt);                        // publish next buffer
#endif
    __syncthreads();
  }

  // LoRA rank-32 correction: one extra K=32 WMMA step straight from global
  // (lowA/lowB are tiny and L2-resident). B concatenates [common | expert(e)].
  {
    Frag af[2], bf4[4];
    #pragma unroll
    for (int tm = 0; tm < 2; ++tm) {
      long r = bm + wm * 32 + tm * 16 + lrow;
      af[tm].q[0] = *reinterpret_cast<const uint4*>(lowA + r * 32 + lk);
      af[tm].q[1] = *reinterpret_cast<const uint4*>(lowA + r * 32 + lk + 16);
    }
    #pragma unroll
    for (int tn = 0; tn < 4; ++tn) {
      long n = bn + wn * 64 + tn * 16 + lrow;
      bf4[tn].q[0] = *reinterpret_cast<const uint4*>(lowBc + n * 16 + lk);                 // K 0-15
      bf4[tn].q[1] = *reinterpret_cast<const uint4*>(lowBe + ((long)e * N + n) * 16 + lk); // K 16-31
    }
    #pragma unroll
    for (int tm = 0; tm < 2; ++tm)
      #pragma unroll
      for (int tn = 0; tn < 4; ++tn)
        acc[tm][tn] = __builtin_amdgcn_wmma_f32_16x16x32_bf16(
            false, af[tm].v, false, bf4[tn].v, (short)0, acc[tm][tn],
            false, false);
  }

  // Epilogue. C layout: VGPR v -> M=v (lanes 0-15) / M=v+8 (lanes 16-31), N=lane%16.
  const int crow = (lane >> 4) * 8;
  const int ccol = lane & 15;
  #pragma unroll
  for (int tm = 0; tm < 2; ++tm) {
    #pragma unroll
    for (int tn = 0; tn < 4; ++tn) {
      int row0 = bm + wm * 32 + tm * 16 + crow;
      int col0 = bn + wn * 64 + tn * 16 + ccol;
      #pragma unroll
      for (int v = 0; v < 8; ++v) {
        float val = acc[tm][tn][v];
        if (RELU) val = val > 0.f ? val : 0.f;
        if constexpr (sizeof(OutT) == 2)
          C[(long)(row0 + v) * N + col0] = f2bf(val);
        else
          C[(long)(row0 + v) * N + col0] = val;
      }
    }
  }
}

// ---------------------------------------------------------------------------
extern "C" void kernel_launch(void* const* d_in, const int* in_sizes, int n_in,
                              void* d_out, int out_size, void* d_ws, size_t ws_size,
                              hipStream_t stream) {
  const float* x    = (const float*)d_in[0];
  const int*   eids = (const int*)  d_in[1];
  const float* wi   = (const float*)d_in[2];
  const float* wo   = (const float*)d_in[3];
  const float* cwia = (const float*)d_in[4];
  const float* cwib = (const float*)d_in[5];
  const float* cwoa = (const float*)d_in[6];
  const float* cwob = (const float*)d_in[7];
  const float* ewia = (const float*)d_in[8];
  const float* ewib = (const float*)d_in[9];
  const float* ewoa = (const float*)d_in[10];
  const float* ewob = (const float*)d_in[11];
  float* out = (float*)d_out;

  // Workspace carve-out (~192 MB total, 256B-aligned slices).
  char* ws = (char*)d_ws;
  size_t off = 0;
  auto carve = [&](size_t bytes) -> char* {
    char* p = ws + off;
    off = (off + bytes + 255) & ~(size_t)255;
    return p;
  };
  unsigned short* xb    = (unsigned short*)carve((size_t)MROWS_ * DIM_ * 2);
  unsigned short* wib   = (unsigned short*)carve((size_t)FF_ * DIM_ * 2);
  unsigned short* wob   = (unsigned short*)carve((size_t)DIM_ * FF_ * 2);
  unsigned short* hb    = (unsigned short*)carve((size_t)MROWS_ * FF_ * 2);
  unsigned short* low1  = (unsigned short*)carve((size_t)MROWS_ * 32 * 2);
  unsigned short* low2  = (unsigned short*)carve((size_t)MROWS_ * 32 * 2);
  unsigned short* cwiab = (unsigned short*)carve((size_t)16 * DIM_ * 2);
  unsigned short* ewiab = (unsigned short*)carve((size_t)NEXP_ * 16 * DIM_ * 2);
  unsigned short* cwoab = (unsigned short*)carve((size_t)16 * FF_ * 2);
  unsigned short* ewoab = (unsigned short*)carve((size_t)NEXP_ * 16 * FF_ * 2);
  unsigned short* cwibb = (unsigned short*)carve((size_t)FF_ * 16 * 2);
  unsigned short* ewibb = (unsigned short*)carve((size_t)NEXP_ * FF_ * 16 * 2);
  unsigned short* cwobb = (unsigned short*)carve((size_t)DIM_ * 16 * 2);
  unsigned short* ewobb = (unsigned short*)carve((size_t)NEXP_ * DIM_ * 16 * 2);

  auto cvt = [&](const float* in, unsigned short* o, long n) {
    long blocks = (n + 255) / 256;
    if (blocks > 4096) blocks = 4096;
    cvt_bf16_kernel<<<dim3((unsigned)blocks), dim3(256), 0, stream>>>(in, o, n);
  };
  cvt(x,    xb,    (long)MROWS_ * DIM_);
  cvt(wi,   wib,   (long)FF_ * DIM_);
  cvt(wo,   wob,   (long)DIM_ * FF_);
  cvt(cwia, cwiab, (long)16 * DIM_);
  cvt(ewia, ewiab, (long)NEXP_ * 16 * DIM_);
  cvt(cwoa, cwoab, (long)16 * FF_);
  cvt(ewoa, ewoab, (long)NEXP_ * 16 * FF_);
  cvt(cwib, cwibb, (long)FF_ * 16);
  cvt(ewib, ewibb, (long)NEXP_ * FF_ * 16);
  cvt(cwob, cwobb, (long)DIM_ * 16);
  cvt(ewob, ewobb, (long)NEXP_ * DIM_ * 16);

  // Layer 1: rank-32 activations (WMMA) then fused GEMM + LoRA + ReLU -> hb.
  lowrank_wmma_kernel<<<dim3(MROWS_ / 128), dim3(256), 0, stream>>>(
      xb, cwiab, ewiab, eids, low1, DIM_);
  gemm_lora_kernel<true, unsigned short>
      <<<dim3(FF_ / BN, MROWS_ / BM), dim3(256), 0, stream>>>(
          xb, wib, low1, cwibb, ewibb, eids, hb, MROWS_, FF_, DIM_);

  // Layer 2: rank-32 on hidden (WMMA), fused GEMM + LoRA -> f32 output.
  lowrank_wmma_kernel<<<dim3(MROWS_ / 128), dim3(256), 0, stream>>>(
      hb, cwoab, ewoab, eids, low2, FF_);
  gemm_lora_kernel<false, float>
      <<<dim3(DIM_ / BN, MROWS_ / BM), dim3(256), 0, stream>>>(
          hb, wob, low2, cwobb, ewobb, eids, out, MROWS_, DIM_, FF_);
}
